// ANFM_84851373899958
// MI455X (gfx1250) — matrix operand order repaired and checked by reading, not compiled, not run on hardware
//
#include <hip/hip_runtime.h>
#include <hip/hip_bf16.h>

#define B_F 26
#define V_N 20000
#define E_N 64
#define P_N 325
#define P_PAD 336   // 21 * 16
#define MT_N 21

typedef __attribute__((ext_vector_type(16))) __bf16 v16bf;
typedef __attribute__((ext_vector_type(8)))  float  v8f;

union FragBF {
    v16bf v;
    unsigned int u[8];
};

__device__ __forceinline__ unsigned short f32_to_bf16(float f) {
    unsigned int x = __float_as_uint(f);
    unsigned int r = x + 0x7FFFu + ((x >> 16) & 1u);   // round-to-nearest-even
    return (unsigned short)(r >> 16);
}
__device__ __forceinline__ float bf16_to_f32(unsigned short u) {
    return __uint_as_float(((unsigned int)u) << 16);
}

// One workgroup (256 threads = 8 wave32) per batch element.
// Fused: gather -> pairs(bf16, LDS) -> WMMA attention GEMM -> score/softmax -> afm.
__global__ __launch_bounds__(256)
void afm_fused_kernel(const int*   __restrict__ x_idx,
                      const float* __restrict__ embed_w,
                      const float* __restrict__ embed_b,
                      const float* __restrict__ att_w,
                      const float* __restrict__ att_b,
                      const float* __restrict__ att_p,
                      float* __restrict__ g_afm,
                      float* __restrict__ g_lr)
{
    __shared__ float          s_xw[B_F * E_N];          // gathered embeddings, fp32
    __shared__ unsigned short s_pairs[P_PAD * E_N];     // pair products, bf16
    __shared__ unsigned short s_wT[E_N * E_N];          // attention_w transposed [n][k], bf16
    __shared__ float          s_ab[E_N];                // attention bias
    __shared__ float          s_ap[E_N];                // attention projection p
    __shared__ float          s_scores[P_PAD];
    __shared__ float          s_part[4 * E_N];
    __shared__ unsigned char  s_pi[P_PAD];
    __shared__ unsigned char  s_pj[P_PAD];
    __shared__ float          s_lr[B_F];
    __shared__ float          s_sum;

    const int tid  = threadIdx.x;
    const int b    = blockIdx.x;
    const int lane = tid & 31;
    const int wid  = tid >> 5;

    // ---- phase 0: gathers (coalesced 64-wide rows; table is L2-resident) ----
    for (int u = tid; u < B_F * E_N; u += 256) {
        int f = u >> 6;
        int e = u & 63;
        int row = x_idx[b * B_F + f];
        s_xw[u] = embed_w[((size_t)f * V_N + row) * E_N + e];
    }
    if (tid < B_F) {
        int row = x_idx[b * B_F + tid];
        s_lr[tid] = embed_b[(size_t)tid * V_N + row];
    }
    if (tid < E_N) {
        s_ab[tid] = att_b[tid];
        s_ap[tid] = att_p[tid];
    }
    // attention_w [e][f] fp32 -> LDS bf16 transposed [f][e] so (k,k+1) pairs are contiguous
    for (int u = tid; u < E_N * E_N; u += 256) {
        int e = u >> 6;
        int f = u & 63;
        s_wT[f * E_N + e] = f32_to_bf16(att_w[u]);
    }
    // triu(k=1) pair-index table, row-major (i<j) order
    if (tid < P_N) {
        int rem = tid, i = 0;
        while (rem >= (B_F - 1 - i)) { rem -= (B_F - 1 - i); ++i; }
        s_pi[tid] = (unsigned char)i;
        s_pj[tid] = (unsigned char)(i + 1 + rem);
    }
    __syncthreads();

    // ---- phase 1: pairwise products -> bf16 in LDS (rows >= 325 zeroed) ----
    for (int u = tid; u < P_PAD * E_N; u += 256) {
        int p = u >> 6;
        int e = u & 63;
        float v = 0.0f;
        if (p < P_N) v = s_xw[s_pi[p] * E_N + e] * s_xw[s_pj[p] * E_N + e];
        s_pairs[u] = f32_to_bf16(v);
    }
    __syncthreads();

    // ---- phase 2: WMMA GEMM (pairs[336x64] @ W[64x64]) fused with ReLU*p row-reduce ----
    {
        const int h = lane >> 4;   // half-wave select per ISA operand layout
        const int m = lane & 15;   // A row within tile
        const int n = lane & 15;   // B/C column within tile

        for (int mt = wid; mt < MT_N; mt += 8) {
            // A fragments: 16x32 bf16, lane m = row, VGPR v holds packed (K, K+1)
            FragBF a0, a1;
            const int rowbase = (mt * 16 + m) * E_N;
            #pragma unroll
            for (int v = 0; v < 8; ++v) {
                int k = (v < 4) ? (h * 8 + 2 * v) : (16 + h * 8 + 2 * (v - 4));
                a0.u[v] = *(const unsigned int*)&s_pairs[rowbase + k];        // K = 0..31
                a1.u[v] = *(const unsigned int*)&s_pairs[rowbase + 32 + k];   // K = 32..63
            }

            float sacc[8];
            #pragma unroll
            for (int v = 0; v < 8; ++v) sacc[v] = 0.0f;

            for (int nt = 0; nt < 4; ++nt) {
                // B fragments: 32x16 bf16, lanes 0-15 K=0..7,16..23, lanes 16-31 K=8..15,24..31
                FragBF b0f, b1f;
                const int nb = (nt * 16 + n) * E_N;
                #pragma unroll
                for (int v = 0; v < 8; ++v) {
                    int k = (v < 4) ? (h * 8 + 2 * v) : (16 + h * 8 + 2 * (v - 4));
                    b0f.u[v] = *(const unsigned int*)&s_wT[nb + k];
                    b1f.u[v] = *(const unsigned int*)&s_wT[nb + 32 + k];
                }
                // accumulator initialized with the broadcast attention bias b[n]
                float biasn = s_ab[nt * 16 + n];
                v8f c;
                #pragma unroll
                for (int v = 0; v < 8; ++v) c[v] = biasn;

                c = __builtin_amdgcn_wmma_f32_16x16x32_bf16(
                        false, a0.v, false, b0f.v, (short)0, c, false, false);
                c = __builtin_amdgcn_wmma_f32_16x16x32_bf16(
                        false, a1.v, false, b1f.v, (short)0, c, false, false);

                // score contribution: sum_n p[n] * relu(att_h[m][n])
                float pw = s_ap[nt * 16 + n];
                #pragma unroll
                for (int v = 0; v < 8; ++v)
                    sacc[v] += fmaxf(c[v], 0.0f) * pw;
            }
            // reduce over the 16 columns (xor masks < 16 stay inside each half-wave)
            #pragma unroll
            for (int v = 0; v < 8; ++v) {
                float x = sacc[v];
                x += __shfl_xor(x, 1, 32);
                x += __shfl_xor(x, 2, 32);
                x += __shfl_xor(x, 4, 32);
                x += __shfl_xor(x, 8, 32);
                sacc[v] = x;
            }
            if ((lane & 15) == 0) {
                #pragma unroll
                for (int v = 0; v < 8; ++v)
                    s_scores[mt * 16 + h * 8 + v] = sacc[v];   // row v (lane0) / v+8 (lane16)
            }
        }
    }
    __syncthreads();

    // ---- phase 3: softmax over 325 scores (wave 0) ; LR reduce (thread 32) ----
    if (tid < 32) {
        float mx = -3.0e38f;
        for (int p = lane; p < P_N; p += 32) mx = fmaxf(mx, s_scores[p]);
        mx = fmaxf(mx, __shfl_xor(mx, 1, 32));
        mx = fmaxf(mx, __shfl_xor(mx, 2, 32));
        mx = fmaxf(mx, __shfl_xor(mx, 4, 32));
        mx = fmaxf(mx, __shfl_xor(mx, 8, 32));
        mx = fmaxf(mx, __shfl_xor(mx, 16, 32));
        float sm = 0.0f;
        for (int p = lane; p < P_N; p += 32) {
            float e = __expf(s_scores[p] - mx);
            s_scores[p] = e;           // store unnormalized weights
            sm += e;
        }
        sm += __shfl_xor(sm, 1, 32);
        sm += __shfl_xor(sm, 2, 32);
        sm += __shfl_xor(sm, 4, 32);
        sm += __shfl_xor(sm, 8, 32);
        sm += __shfl_xor(sm, 16, 32);
        if (lane == 0) s_sum = sm;
    }
    if (tid == 32) {
        float l = 0.0f;
        #pragma unroll
        for (int f = 0; f < B_F; ++f) l += s_lr[f];
        g_lr[b] = l;
    }
    __syncthreads();

    // ---- phase 4: afm[e] = (1/sum) * sum_p w_p * pairs[p][e] ----
    {
        const float inv = 1.0f / s_sum;
        int e = tid & 63;
        int chunk = tid >> 6;
        float part = 0.0f;
        for (int p = chunk; p < P_N; p += 4)
            part += s_scores[p] * bf16_to_f32(s_pairs[p * E_N + e]);
        s_part[chunk * E_N + e] = part;
        __syncthreads();
        if (tid < E_N) {
            float a = (s_part[tid] + s_part[E_N + tid] +
                       s_part[2 * E_N + tid] + s_part[3 * E_N + tid]) * inv;
            g_afm[(size_t)b * E_N + tid] = a;
        }
    }
}

// MLP head: 16 batch rows per block (64->256->128->1). ~0.4 GFLOP total: plain VALU.
__global__ __launch_bounds__(256)
void afm_mlp_kernel(const float* __restrict__ g_afm,
                    const float* __restrict__ g_lr,
                    const float* __restrict__ w0, const float* __restrict__ b0,
                    const float* __restrict__ w1, const float* __restrict__ b1,
                    const float* __restrict__ w2, const float* __restrict__ b2,
                    const float* __restrict__ bias,
                    float* __restrict__ out)
{
    __shared__ float s_a[16 * 64];
    __shared__ float s_h0[16 * 256];
    __shared__ float s_h1[16 * 128];
    const int tid = threadIdx.x;
    const int rb  = blockIdx.x * 16;

    for (int u = tid; u < 16 * 64; u += 256)
        s_a[u] = g_afm[(size_t)rb * 64 + u];
    __syncthreads();

    // layer 0: each thread owns one of 256 output neurons across all 16 rows
    {
        float acc[16];
        float bv = b0[tid];
        #pragma unroll
        for (int r = 0; r < 16; ++r) acc[r] = bv;
        for (int k = 0; k < 64; ++k) {
            float w = w0[k * 256 + tid];
            #pragma unroll
            for (int r = 0; r < 16; ++r)
                acc[r] = fmaf(s_a[r * 64 + k], w, acc[r]);
        }
        #pragma unroll
        for (int r = 0; r < 16; ++r)
            s_h0[r * 256 + tid] = fmaxf(acc[r], 0.0f);
    }
    __syncthreads();

    // layer 1: thread -> (neuron n = tid%128, row half rh = tid/128 -> 8 rows)
    {
        int n  = tid & 127;
        int rh = tid >> 7;
        float acc[8];
        float bv = b1[n];
        #pragma unroll
        for (int r = 0; r < 8; ++r) acc[r] = bv;
        for (int k = 0; k < 256; ++k) {
            float w = w1[k * 128 + n];
            #pragma unroll
            for (int r = 0; r < 8; ++r)
                acc[r] = fmaf(s_h0[(rh * 8 + r) * 256 + k], w, acc[r]);
        }
        #pragma unroll
        for (int r = 0; r < 8; ++r)
            s_h1[(rh * 8 + r) * 128 + n] = fmaxf(acc[r], 0.0f);
    }
    __syncthreads();

    // output: 16 lanes per row dot with w2[128], shfl reduce, add LR + biases
    {
        int row = tid >> 4;
        int l   = tid & 15;
        float p = 0.0f;
        for (int k = l; k < 128; k += 16)
            p = fmaf(s_h1[row * 128 + k], w2[k], p);
        p += __shfl_xor(p, 1, 32);
        p += __shfl_xor(p, 2, 32);
        p += __shfl_xor(p, 4, 32);
        p += __shfl_xor(p, 8, 32);
        if (l == 0)
            out[rb + row] = p + b2[0] + g_lr[rb + row] + bias[0];
    }
}

extern "C" void kernel_launch(void* const* d_in, const int* in_sizes, int n_in,
                              void* d_out, int out_size, void* d_ws, size_t ws_size,
                              hipStream_t stream)
{
    const int*   x_idx   = (const int*)  d_in[0];
    const float* embed_w = (const float*)d_in[1];
    const float* embed_b = (const float*)d_in[2];
    const float* att_w   = (const float*)d_in[3];
    const float* att_b   = (const float*)d_in[4];
    const float* att_p   = (const float*)d_in[5];
    const float* w0      = (const float*)d_in[6];
    const float* b0      = (const float*)d_in[7];
    const float* w1      = (const float*)d_in[8];
    const float* b1      = (const float*)d_in[9];
    const float* w2      = (const float*)d_in[10];
    const float* b2      = (const float*)d_in[11];
    const float* bias    = (const float*)d_in[12];
    float* out = (float*)d_out;

    const int B = in_sizes[0] / B_F;           // 4096

    float* g_afm = (float*)d_ws;               // [B, 64]
    float* g_lr  = g_afm + (size_t)B * E_N;    // [B]

    afm_fused_kernel<<<B, 256, 0, stream>>>(x_idx, embed_w, embed_b,
                                            att_w, att_b, att_p, g_afm, g_lr);
    afm_mlp_kernel<<<B / 16, 256, 0, stream>>>(g_afm, g_lr,
                                               w0, b0, w1, b1, w2, b2, bias, out);
}
